// KimiSparseMoeBlock_75823352644316
// MI455X (gfx1250) — compile-verified, matrix operand order checked by reading, not run on hardware
//
#include <hip/hip_runtime.h>

// ---------------- problem constants (match reference) ----------------
#define T_TOK  2048
#define HDIM   1024
#define FDIM   512
#define F2DIM  1024
#define EEXP   64
#define TOPK   8
#define NGRP   8
#define TGRP   4
#define CAPCT  1024
#define SCALEF 2.5f
#define SLOTS  (T_TOK * TOPK)

// ---------------- WMMA types (gfx1250, wave32) ----------------
typedef __attribute__((ext_vector_type(16))) __bf16 v16bf;
typedef __attribute__((ext_vector_type(2)))  __bf16 v2bf;
typedef __attribute__((ext_vector_type(8)))  float  v8f;

union Frag { v16bf v; uint4 q[2]; };

__device__ __forceinline__ unsigned short f2bf(float f) {
  // round-to-nearest-even fp32 -> bf16 (raw bits), scalar fallback
  unsigned int u = __float_as_uint(f);
  u += 0x7fffu + ((u >> 16) & 1u);
  return (unsigned short)(u >> 16);
}

__device__ __forceinline__ unsigned int pack_bf16(float lo, float hi) {
#if __has_builtin(__builtin_amdgcn_cvt_pk_bf16_f32)
  v2bf r = __builtin_amdgcn_cvt_pk_bf16_f32(lo, hi);
  return __builtin_bit_cast(unsigned int, r);
#else
  // gfx1250 V_CVT_PK_BF16_F32: 2 x f32 -> packed 2 x bf16, RNE, one VALU op
  unsigned int r;
  asm("v_cvt_pk_bf16_f32 %0, %1, %2" : "=v"(r) : "v"(lo), "v"(hi));
  return r;
#endif
}

// ---------------- small utility kernels ----------------
__global__ void zero_ints_kernel(int* p, int n) {
  int i = blockIdx.x * blockDim.x + threadIdx.x;
  if (i < n) p[i] = 0;
}

// packed fp32 -> bf16 conversion, 2 elements per thread iteration
__global__ void convert_bf16_kernel(const float* __restrict__ src,
                                    unsigned int* __restrict__ dst, int npairs) {
  int i = blockIdx.x * blockDim.x + threadIdx.x;
  int stride = gridDim.x * blockDim.x;
  for (; i < npairs; i += stride) {
    float2 f = ((const float2*)src)[i];
    dst[i] = pack_bf16(f.x, f.y);
  }
}

// ---------------- routing: one block (256 thr) per token ----------------
__global__ __launch_bounds__(256)
void route_kernel(const float* __restrict__ x, const float* __restrict__ gw,
                  const float* __restrict__ ebias,
                  int* __restrict__ topk_idx, float* __restrict__ topk_w,
                  int* __restrict__ counts) {
  const int t = blockIdx.x;
  const int tid = threadIdx.x;
  __shared__ float part[256];
  __shared__ float sc[EEXP];   // sigmoid scores
  __shared__ float sfc[EEXP];  // scores + bias
  __shared__ float gsc[NGRP];

  const int e = tid & (EEXP - 1);
  const int chunk = tid >> 6;  // 0..3, each 256 elements of H
  const float* xr = x + (size_t)t * HDIM + chunk * (HDIM / 4);
  const float* wr = gw + (size_t)e * HDIM + chunk * (HDIM / 4);
  float s = 0.f;
#pragma unroll 8
  for (int j = 0; j < HDIM / 4; ++j) s += xr[j] * wr[j];
  part[tid] = s;
  __syncthreads();

  if (tid < EEXP) {
    float l = part[tid] + part[tid + 64] + part[tid + 128] + part[tid + 192];
    float sg = 1.f / (1.f + expf(-l));
    sc[tid] = sg;
    sfc[tid] = sg + ebias[tid];
  }
  __syncthreads();

  if (tid < NGRP) {  // sum of top-2 sfc per group of 8
    float m1 = -1e30f, m2 = -1e30f;
    for (int j = 0; j < EEXP / NGRP; ++j) {
      float v = sfc[tid * (EEXP / NGRP) + j];
      if (v > m1) { m2 = m1; m1 = v; }
      else if (v > m2) m2 = v;
    }
    gsc[tid] = m1 + m2;
  }
  __syncthreads();

  if (tid == 0) {
    unsigned gmask = 0;
    for (int r = 0; r < TGRP; ++r) {
      int best = 0; float bv = -1e30f;
      for (int g = 0; g < NGRP; ++g)
        if (!((gmask >> g) & 1) && gsc[g] > bv) { bv = gsc[g]; best = g; }
      gmask |= 1u << best;
    }
    int idxs[TOPK]; float ws[TOPK]; float wsum = 0.f;
    unsigned long long used = 0ull;
    for (int r = 0; r < TOPK; ++r) {
      int best = 0; float bv = -1e30f;
      for (int ee = 0; ee < EEXP; ++ee) {
        if ((used >> ee) & 1ull) continue;
        if (!((gmask >> (ee / (EEXP / NGRP))) & 1)) continue;
        float v = sfc[ee];
        if (v > bv) { bv = v; best = ee; }
      }
      used |= 1ull << best;
      idxs[r] = best; ws[r] = sc[best]; wsum += sc[best];
    }
    float inv = SCALEF / (wsum + 1e-20f);
    for (int r = 0; r < TOPK; ++r) {
      topk_idx[t * TOPK + r] = idxs[r];
      topk_w[t * TOPK + r] = ws[r] * inv;
      atomicAdd(&counts[idxs[r]], 1);
    }
  }
}

__global__ void offsets_kernel(const int* __restrict__ counts,
                               int* __restrict__ offs, int* __restrict__ clamped) {
  if (threadIdx.x == 0 && blockIdx.x == 0) {
    int off = 0;
    for (int e2 = 0; e2 < EEXP; ++e2) {
      offs[e2] = off;
      off += counts[e2];
      clamped[e2] = counts[e2] < CAPCT ? counts[e2] : CAPCT;
    }
  }
}

__global__ void scatter_kernel(const int* __restrict__ topk_idx,
                               const float* __restrict__ topk_w,
                               const int* __restrict__ offs, int* __restrict__ fill,
                               int* __restrict__ slot_tok, float* __restrict__ slot_w) {
  int id = blockIdx.x * blockDim.x + threadIdx.x;
  if (id >= SLOTS) return;
  int e2 = topk_idx[id];
  int p = atomicAdd(&fill[e2], 1);
  if (p < CAPCT) {
    int r = offs[e2] + p;
    slot_tok[r] = id / TOPK;
    slot_w[r] = topk_w[id];
  }
}

// ---------------- WMMA GEMM kernels ----------------
// LDS row stride in bf16 elements: 32 data + 8 pad (80B rows -> conflict-free)
#define LDA 40

// Hout[row, n] = silu(A @ Wg^T) * (A @ Wu^T), bf16 out.
// A rows from Abf (bf16 [*,Kdim]) via rowIdx indirection (or identity).
// Per-expert via blockIdx.z; shared path: cnt/offs/rowIdx null, Mfixed rows.
__global__ __launch_bounds__(256)
void gemm_gateup_kernel(const unsigned short* __restrict__ Abf,
                        const int* __restrict__ rowIdx,
                        const int* __restrict__ cnt, const int* __restrict__ offs,
                        const float* __restrict__ WgBase, const float* __restrict__ WuBase,
                        long long wstride,
                        unsigned short* __restrict__ Hout,
                        int Ntot, int Kdim, int Mfixed) {
  const int mtile = blockIdx.x, ntile = blockIdx.y, eidx = blockIdx.z;
  const int count = cnt ? cnt[eidx] : Mfixed;
  if (mtile * 128 >= count) return;
  const int rbase = offs ? offs[eidx] : 0;
  const float* Wg = WgBase + (long long)eidx * wstride;
  const float* Wu = WuBase + (long long)eidx * wstride;

  __shared__ __align__(16) unsigned short As[2][128 * LDA];
  __shared__ __align__(16) unsigned short Bg[2][64 * LDA];
  __shared__ __align__(16) unsigned short Bu[2][64 * LDA];
  __shared__ int tokRow[128];

  const int tid = threadIdx.x;
  if (tid < 128) {
    int rl = mtile * 128 + tid;
    tokRow[tid] = (rl < count) ? (rowIdx ? rowIdx[rbase + rl] : rl) : -1;
  }
  __syncthreads();

  // per-thread tile-load coordinates
  const int arow = tid >> 1;                 // 0..127
  const int akof = (tid & 1) << 4;           // 0 or 16
  const int atok = tokRow[arow];
  const int brow = tid >> 2;                 // 0..63
  const int bkof = (tid & 3) << 3;           // 0,8,16,24
  const size_t bofs = (size_t)(ntile * 64 + brow) * Kdim + bkof;

  // fragment coordinates
  const int wid = tid >> 5, lane = tid & 31;
  const int nrow = lane & 15;
  const int khalf = (lane < 16) ? 0 : 1;

  v8f zero8;
#pragma unroll
  for (int i = 0; i < 8; ++i) zero8[i] = 0.f;
  v8f accG[4], accU[4];
#pragma unroll
  for (int i = 0; i < 4; ++i) { accG[i] = zero8; accU[i] = zero8; }

  // ---- prologue: load K-tile 0 into buffer 0 ----
  uint4 ad0 = make_uint4(0, 0, 0, 0), ad1 = ad0;
  float bg[8], bu[8];
  if (atok >= 0) {
    const uint4* src = (const uint4*)(Abf + (size_t)atok * Kdim + akof);
    ad0 = src[0]; ad1 = src[1];
  }
#pragma unroll
  for (int j = 0; j < 8; ++j) { bg[j] = Wg[bofs + j]; bu[j] = Wu[bofs + j]; }
  {
    uint4* ad = (uint4*)(As[0] + arow * LDA + akof);
    ad[0] = ad0; ad[1] = ad1;
    *(uint4*)(Bg[0] + brow * LDA + bkof) =
        make_uint4(pack_bf16(bg[0], bg[1]), pack_bf16(bg[2], bg[3]),
                   pack_bf16(bg[4], bg[5]), pack_bf16(bg[6], bg[7]));
    *(uint4*)(Bu[0] + brow * LDA + bkof) =
        make_uint4(pack_bf16(bu[0], bu[1]), pack_bf16(bu[2], bu[3]),
                   pack_bf16(bu[4], bu[5]), pack_bf16(bu[6], bu[7]));
  }
  __syncthreads();

  int p = 0;
  for (int kk = 0; kk < Kdim; kk += 32) {
    const bool hasNext = (kk + 32) < Kdim;
    // ---- prefetch next K-tile into registers (overlaps WMMAs below) ----
    if (hasNext) {
      ad0 = make_uint4(0, 0, 0, 0); ad1 = ad0;
      if (atok >= 0) {
        const uint4* src = (const uint4*)(Abf + (size_t)atok * Kdim + kk + 32 + akof);
        ad0 = src[0]; ad1 = src[1];
      }
#pragma unroll
      for (int j = 0; j < 8; ++j) {
        bg[j] = Wg[bofs + kk + 32 + j];
        bu[j] = Wu[bofs + kk + 32 + j];
      }
    }
    // ---- compute from LDS buffer p ----
    Frag fa;
    const uint4* ar = (const uint4*)(As[p] + (wid * 16 + nrow) * LDA);
    fa.q[0] = ar[khalf];
    fa.q[1] = ar[2 + khalf];
#pragma unroll
    for (int ns = 0; ns < 4; ++ns) {
      Frag fbg, fbu;
      const uint4* bgr = (const uint4*)(Bg[p] + (ns * 16 + nrow) * LDA);
      const uint4* bur = (const uint4*)(Bu[p] + (ns * 16 + nrow) * LDA);
      fbg.q[0] = bgr[khalf]; fbg.q[1] = bgr[2 + khalf];
      fbu.q[0] = bur[khalf]; fbu.q[1] = bur[2 + khalf];
      accG[ns] = __builtin_amdgcn_wmma_f32_16x16x32_bf16(
          false, fa.v, false, fbg.v, (short)0, accG[ns], false, false);
      accU[ns] = __builtin_amdgcn_wmma_f32_16x16x32_bf16(
          false, fa.v, false, fbu.v, (short)0, accU[ns], false, false);
    }
    // ---- stage next tile into the other buffer ----
    if (hasNext) {
      uint4* ad = (uint4*)(As[p ^ 1] + arow * LDA + akof);
      ad[0] = ad0; ad[1] = ad1;
      *(uint4*)(Bg[p ^ 1] + brow * LDA + bkof) =
          make_uint4(pack_bf16(bg[0], bg[1]), pack_bf16(bg[2], bg[3]),
                     pack_bf16(bg[4], bg[5]), pack_bf16(bg[6], bg[7]));
      *(uint4*)(Bu[p ^ 1] + brow * LDA + bkof) =
          make_uint4(pack_bf16(bu[0], bu[1]), pack_bf16(bu[2], bu[3]),
                     pack_bf16(bu[4], bu[5]), pack_bf16(bu[6], bu[7]));
    }
    __syncthreads();
    p ^= 1;
  }

  // epilogue: h = silu(g) * u, bf16 store into compact Hout
  const int mofs = (lane < 16) ? 0 : 8;
#pragma unroll
  for (int ns = 0; ns < 4; ++ns) {
    int col = ntile * 64 + ns * 16 + nrow;
#pragma unroll
    for (int i = 0; i < 8; ++i) {
      int rl = mtile * 128 + wid * 16 + i + mofs;
      if (rl < count) {
        float g = accG[ns][i], u = accU[ns][i];
        float h = g / (1.f + expf(-g)) * u;
        Hout[(size_t)(rbase + rl) * Ntot + col] =
            (unsigned short)(pack_bf16(h, 0.f) & 0xffffu);
      }
    }
  }
}

// out = Hin @ W^T ; expert path: weighted atomic scatter to tokens.
// shared path (slotTok==null): plain store (initializes d_out).
__global__ __launch_bounds__(256)
void gemm_down_kernel(const unsigned short* __restrict__ Hin,
                      const int* __restrict__ cnt, const int* __restrict__ offs,
                      const int* __restrict__ slotTok, const float* __restrict__ slotW,
                      const float* __restrict__ WBase, long long wstride,
                      float* __restrict__ out, int Ntot, int Kdim, int Mfixed) {
  const int mtile = blockIdx.x, ntile = blockIdx.y, eidx = blockIdx.z;
  const int count = cnt ? cnt[eidx] : Mfixed;
  if (mtile * 128 >= count) return;
  const int rbase = offs ? offs[eidx] : 0;
  const float* W = WBase + (long long)eidx * wstride;

  __shared__ __align__(16) unsigned short As[2][128 * LDA];
  __shared__ __align__(16) unsigned short Bs[2][64 * LDA];

  const int tid = threadIdx.x;
  const int arow = tid >> 1;
  const int akof = (tid & 1) << 4;
  const bool avalid = (mtile * 128 + arow) < count;
  const size_t aofs = (size_t)(rbase + mtile * 128 + arow) * Kdim + akof;
  const int brow = tid >> 2;
  const int bkof = (tid & 3) << 3;
  const size_t bofs = (size_t)(ntile * 64 + brow) * Kdim + bkof;

  const int wid = tid >> 5, lane = tid & 31;
  const int nrow = lane & 15;
  const int khalf = (lane < 16) ? 0 : 1;
  v8f zero8;
#pragma unroll
  for (int i = 0; i < 8; ++i) zero8[i] = 0.f;
  v8f acc[4];
#pragma unroll
  for (int i = 0; i < 4; ++i) acc[i] = zero8;

  uint4 ad0 = make_uint4(0, 0, 0, 0), ad1 = ad0;
  float bw[8];
  if (avalid) {
    const uint4* src = (const uint4*)(Hin + aofs);
    ad0 = src[0]; ad1 = src[1];
  }
#pragma unroll
  for (int j = 0; j < 8; ++j) bw[j] = W[bofs + j];
  {
    uint4* ad = (uint4*)(As[0] + arow * LDA + akof);
    ad[0] = ad0; ad[1] = ad1;
    *(uint4*)(Bs[0] + brow * LDA + bkof) =
        make_uint4(pack_bf16(bw[0], bw[1]), pack_bf16(bw[2], bw[3]),
                   pack_bf16(bw[4], bw[5]), pack_bf16(bw[6], bw[7]));
  }
  __syncthreads();

  int p = 0;
  for (int kk = 0; kk < Kdim; kk += 32) {
    const bool hasNext = (kk + 32) < Kdim;
    if (hasNext) {
      ad0 = make_uint4(0, 0, 0, 0); ad1 = ad0;
      if (avalid) {
        const uint4* src = (const uint4*)(Hin + aofs + kk + 32);
        ad0 = src[0]; ad1 = src[1];
      }
#pragma unroll
      for (int j = 0; j < 8; ++j) bw[j] = W[bofs + kk + 32 + j];
    }
    Frag fa;
    const uint4* ar = (const uint4*)(As[p] + (wid * 16 + nrow) * LDA);
    fa.q[0] = ar[khalf];
    fa.q[1] = ar[2 + khalf];
#pragma unroll
    for (int ns = 0; ns < 4; ++ns) {
      Frag fb;
      const uint4* br = (const uint4*)(Bs[p] + (ns * 16 + nrow) * LDA);
      fb.q[0] = br[khalf]; fb.q[1] = br[2 + khalf];
      acc[ns] = __builtin_amdgcn_wmma_f32_16x16x32_bf16(
          false, fa.v, false, fb.v, (short)0, acc[ns], false, false);
    }
    if (hasNext) {
      uint4* ad = (uint4*)(As[p ^ 1] + arow * LDA + akof);
      ad[0] = ad0; ad[1] = ad1;
      *(uint4*)(Bs[p ^ 1] + brow * LDA + bkof) =
          make_uint4(pack_bf16(bw[0], bw[1]), pack_bf16(bw[2], bw[3]),
                     pack_bf16(bw[4], bw[5]), pack_bf16(bw[6], bw[7]));
    }
    __syncthreads();
    p ^= 1;
  }

  const bool atom = (slotTok != nullptr);
  const int mofs = (lane < 16) ? 0 : 8;
#pragma unroll
  for (int ns = 0; ns < 4; ++ns) {
    int col = ntile * 64 + ns * 16 + nrow;
#pragma unroll
    for (int i = 0; i < 8; ++i) {
      int rl = mtile * 128 + wid * 16 + i + mofs;
      if (rl < count) {
        float v = acc[ns][i];
        if (atom) {
          int rg = rbase + rl;
          atomicAdd(&out[(size_t)slotTok[rg] * Ntot + col], slotW[rg] * v);
        } else {
          out[(size_t)rl * Ntot + col] = v;
        }
      }
    }
  }
}

// ---------------- host-side launch ----------------
extern "C" void kernel_launch(void* const* d_in, const int* in_sizes, int n_in,
                              void* d_out, int out_size, void* d_ws, size_t ws_size,
                              hipStream_t stream) {
  const float* x          = (const float*)d_in[0];  // [1,2048,1024]
  const float* gw         = (const float*)d_in[1];  // [64,1024]
  const float* ebias      = (const float*)d_in[2];  // [64]
  const float* w1         = (const float*)d_in[3];  // [64,512,1024]
  const float* w2         = (const float*)d_in[4];  // [64,1024,512]
  const float* w3         = (const float*)d_in[5];  // [64,512,1024]
  const float* sh_gate    = (const float*)d_in[6];  // [1024,1024]
  const float* sh_up      = (const float*)d_in[7];  // [1024,1024]
  const float* sh_down    = (const float*)d_in[8];  // [1024,1024]
  float* out = (float*)d_out;                       // [1,2048,1024] f32

  // workspace carve-up (256B aligned)
  char* ws = (char*)d_ws;
  size_t off = 0;
  auto alloc = [&](size_t bytes) -> char* {
    char* p = ws + off;
    off = (off + bytes + 255) & ~(size_t)255;
    return p;
  };
  unsigned short* xb  = (unsigned short*)alloc((size_t)T_TOK * HDIM * 2);
  unsigned short* Hh  = (unsigned short*)alloc((size_t)SLOTS * FDIM * 2);
  unsigned short* Hs  = (unsigned short*)alloc((size_t)T_TOK * F2DIM * 2);
  int*   topk_idx = (int*)alloc((size_t)SLOTS * 4);
  float* topk_w   = (float*)alloc((size_t)SLOTS * 4);
  int*   counts   = (int*)alloc((size_t)EEXP * 4);
  int*   fill     = (int*)alloc((size_t)EEXP * 4);
  int*   offs     = (int*)alloc((size_t)EEXP * 4);
  int*   clamped  = (int*)alloc((size_t)EEXP * 4);
  int*   slot_tok = (int*)alloc((size_t)SLOTS * 4);
  float* slot_w   = (float*)alloc((size_t)SLOTS * 4);
  if (off > ws_size) return;  // insufficient workspace; no-op (deterministic)

  // 1. zero per-expert counters
  zero_ints_kernel<<<1, 256, 0, stream>>>(counts, EEXP);
  zero_ints_kernel<<<1, 256, 0, stream>>>(fill, EEXP);

  // 2. activations fp32 -> bf16 (packed, 2 per thread-iter)
  convert_bf16_kernel<<<1024, 256, 0, stream>>>(x, (unsigned int*)xb,
                                                T_TOK * HDIM / 2);

  // 3. routing
  route_kernel<<<T_TOK, 256, 0, stream>>>(x, gw, ebias, topk_idx, topk_w, counts);

  // 4. prefix offsets + clamp to CAP
  offsets_kernel<<<1, 64, 0, stream>>>(counts, offs, clamped);

  // 5. slot scatter (compact row lists)
  scatter_kernel<<<SLOTS / 256, 256, 0, stream>>>(topk_idx, topk_w, offs, fill,
                                                  slot_tok, slot_w);

  // 6. expert gate/up: Hh = silu(Xe@w1^T) * (Xe@w3^T)   [compact, FDIM]
  {
    dim3 grid(CAPCT / 128, FDIM / 64, EEXP);
    gemm_gateup_kernel<<<grid, 256, 0, stream>>>(
        xb, slot_tok, clamped, offs, w1, w3,
        (long long)FDIM * HDIM, Hh, FDIM, HDIM, 0);
  }

  // 7. shared gate/up: Hs = silu(X@sg^T) * (X@su^T)    [T, F2]
  {
    dim3 grid(T_TOK / 128, F2DIM / 64, 1);
    gemm_gateup_kernel<<<grid, 256, 0, stream>>>(
        xb, nullptr, nullptr, nullptr, sh_gate, sh_up,
        0, Hs, F2DIM, HDIM, T_TOK);
  }

  // 8. shared down: out = Hs @ sd^T (plain store, initializes d_out)
  {
    dim3 grid(T_TOK / 128, HDIM / 64, 1);
    gemm_down_kernel<<<grid, 256, 0, stream>>>(
        Hs, nullptr, nullptr, nullptr, nullptr, sh_down,
        0, out, HDIM, F2DIM, T_TOK);
  }

  // 9. expert down: out[tok] += w * (Hh @ w2^T) via f32 atomics
  {
    dim3 grid(CAPCT / 128, HDIM / 64, EEXP);
    gemm_down_kernel<<<grid, 256, 0, stream>>>(
        Hh, clamped, offs, slot_tok, slot_w, w2,
        (long long)HDIM * FDIM, out, HDIM, FDIM, 0);
  }
}
